// Dilated_res_block_635655160059
// MI455X (gfx1250) — matrix operand order repaired and checked by reading, not compiled, not run on hardware
//
#include <hip/hip_runtime.h>

#define BB 4
#define NN 40960
#define KNB 16
#define DIN 32
#define DOUT 64
#define DH 32
#define EPSF 1e-5f
#define SLOPEF 0.01f

typedef _Float16 h16;
typedef __attribute__((ext_vector_type(16))) _Float16 v16h;
typedef __attribute__((ext_vector_type(8)))  _Float16 v8h;
typedef __attribute__((ext_vector_type(8)))  float    v8f;

struct P {
  const float* feature; const float* xyz; const int* nidx;
  const float *W1,*b1,*g1,*be1, *W2,*b2,*g2,*be2;
  const float *Wf1,*bf1,*ag1,*abe1, *Wc1,*bc1,*ag2,*abe2;
  const float *W3,*b3,*g3,*be3;
  const float *Wf2,*bf2,*bg1,*bbe1, *Wc2,*bc2,*bg2,*bbe2;
  const float *W4,*b4,*g4,*be4, *W5,*b5,*g5,*be5;
  float* out;
  h16* f_pc;   // [B*N, 32] f16 workspace
  h16* f_agg;  // [B*N, 32] f16 workspace
};

__device__ __forceinline__ float lrelu(float v){ return v >= 0.f ? v : SLOPEF * v; }

__device__ __forceinline__ v8f wmma16(v16h a, v16h b, v8f c){
  // D = A(16x32 f16) * B(32x16 f16) + C(16x16 f32)
  return __builtin_amdgcn_wmma_f32_16x16x32_f16(false, a, false, b, (short)0, c, false, false);
}

// Load an A fragment (16x32 f16) for one 32-wide K chunk from an LDS row-major
// buffer [16][stride].  Lane: m = lane&15, kbase = (lane>>4)*8; element s holds k = 2*kbase + s.
__device__ __forceinline__ v16h a_frag_lds(const h16* buf, int stride, int m, int kbase, int koff){
  v8h lo = *(const v8h*)(buf + m*stride + koff + 2*kbase);
  v8h hh = *(const v8h*)(buf + m*stride + koff + 2*kbase + 8);
  v16h a;
  #pragma unroll
  for (int i = 0; i < 8; ++i){ a[i] = lo[i]; a[i+8] = hh[i]; }
  return a;
}

// ---------------------------------------------------------------------------
// Kernel 1: f_pc = lrelu(bn(x @ W1 + b1)) -> ws f16 [B*N][32]
// One wave handles 16 points (a full 16x32 A tile).
// ---------------------------------------------------------------------------
__global__ __launch_bounds__(256) void k_fpc(P p){
  __shared__ h16 s_w1[DH * DIN];  // column-major [c][k], bn scale folded
  const int tid = threadIdx.x;
  const float RS = rsqrtf(1.0f + EPSF);
  for (int i = tid; i < DH * DIN; i += 256){
    int c = i / DIN, k = i % DIN;
    s_w1[c*DIN + k] = (h16)(p.W1[k*DH + c] * (p.g1[c] * RS));
  }
  __syncthreads();

  const int wave = tid >> 5, lane = tid & 31;
  const int m = lane & 15, hiw = lane >> 4;
  const int kbase = hiw * 8, roff = hiw * 8;
  const long base = ((long)blockIdx.x * 8 + wave) * 16;
  const long pt = base + m;
  const int b = (int)(pt / NN), n = (int)(pt % NN);

  v16h a;
  #pragma unroll
  for (int h = 0; h < 16; ++h){
    int Kk = hiw*16 + h;
    a[h] = (h16)p.feature[((long)b*DIN + Kk)*NN + n];
  }

  #pragma unroll
  for (int nc = 0; nc < 2; ++nc){
    int col = nc*16 + m;
    v16h bw = *(const v16h*)(s_w1 + col*DIN + hiw*16);
    v8f c = {};
    c = wmma16(a, bw, c);
    float s = p.g1[col] * RS;
    float beta = p.b1[col]*s + p.be1[col];
    #pragma unroll
    for (int r = 0; r < 8; ++r){
      long prow = base + roff + r;
      p.f_pc[prow*DH + col] = (h16)lrelu(c[r] + beta);
    }
  }
}

// ---------------------------------------------------------------------------
// Kernel 2: rel-pos encode + f_cat + attention pool 1 -> f_agg ws.
// One wave per point (f_cat is 16x64: k neighbors x channels).
// ---------------------------------------------------------------------------
__global__ __launch_bounds__(256) void k_att1(P p){
  __shared__ h16 s_wf[DOUT * DOUT];   // Wf1 col-major, scale folded (8KB)
  __shared__ h16 s_w2[32 * 32];       // W2 padded 10->32, col-major (2KB)
  __shared__ float s_batt[DOUT];
  __shared__ float s_b2[DH];
  __shared__ h16 s_fcat[8][16 * 64];  // per-wave f_cat (16KB)
  __shared__ float s_agg[8][64];

  const int tid = threadIdx.x;
  const float RS = rsqrtf(1.0f + EPSF);
  for (int i = tid; i < DOUT * DOUT; i += 256){
    int c = i / DOUT, k = i % DOUT;
    s_wf[c*DOUT + k] = (h16)(p.Wf1[k*DOUT + c] * (p.ag1[c] * RS));
  }
  for (int i = tid; i < 32 * 32; i += 256){
    int c = i / 32, k = i % 32;
    float w = (k < 10) ? p.W2[k*DH + c] : 0.f;
    s_w2[c*32 + k] = (h16)(w * (p.g2[c] * RS));
  }
  for (int i = tid; i < DOUT; i += 256) s_batt[i] = p.bf1[i]*(p.ag1[i]*RS) + p.abe1[i];
  for (int i = tid; i < DH;   i += 256) s_b2[i]   = p.b2[i]*(p.g2[i]*RS) + p.be2[i];
  __syncthreads();

  const int wave = tid >> 5, lane = tid & 31;
  const int m = lane & 15, hiw = lane >> 4;
  const int kbase = hiw * 8, roff = hiw * 8;
  const long pt = (long)blockIdx.x * 8 + wave;
  const int b = (int)(pt / NN), n = (int)(pt % NN);

  const int idxm = p.nidx[pt*KNB + m];
  const float* xb = p.xyz + (long)b * 3 * NN;
  float cx = xb[n],    cy = xb[NN + n],    cz = xb[2*NN + n];
  float nx = xb[idxm], ny = xb[NN + idxm], nz = xb[2*NN + idxm];
  float rx = cx - nx, ry = cy - ny, rz = cz - nz;
  float d = sqrtf(rx*rx + ry*ry + rz*rz);
  float f10[10] = {d, rx, ry, rz, cx, cy, cz, nx, ny, nz};

  v16h ax;
  #pragma unroll
  for (int h = 0; h < 16; ++h) ax[h] = (h16)0.f;
  #pragma unroll
  for (int h = 0; h < 16; ++h){ int Kk = hiw*16 + h; if (Kk < 10) ax[h] = (h16)f10[Kk]; }

  h16* fcat = s_fcat[wave];

  // f_xyz_e = lrelu(bn(f_xyz @ W2)) -> fcat cols [32..63]
  #pragma unroll
  for (int nc = 0; nc < 2; ++nc){
    int col = nc*16 + m;
    v16h bw = *(const v16h*)(s_w2 + col*32 + hiw*16);
    v8f c = {};
    c = wmma16(ax, bw, c);
    float beta = s_b2[col];
    #pragma unroll
    for (int r = 0; r < 8; ++r)
      fcat[(roff + r)*64 + 32 + col] = (h16)lrelu(c[r] + beta);
  }

  // f_nb gather -> fcat cols [0..31]  (each lane copies 32B of one row)
  {
    const h16* src = p.f_pc + ((long)b*NN + idxm)*DH + hiw*16;
    v8h a0 = *(const v8h*)src;
    v8h a1 = *(const v8h*)(src + 8);
    *(v8h*)(fcat + m*64 + hiw*16)     = a0;
    *(v8h*)(fcat + m*64 + hiw*16 + 8) = a1;
  }
  __syncthreads();

  // attention: att = f_cat @ Wf1 (bn folded), softmax over k, agg
  #pragma unroll
  for (int nc = 0; nc < 4; ++nc){
    v8f c = {};
    #pragma unroll
    for (int kc = 0; kc < 2; ++kc){
      v16h a = a_frag_lds(fcat, 64, m, kbase, kc*32);
      v16h bw = *(const v16h*)(s_wf + (nc*16 + m)*DOUT + kc*32 + hiw*16);
      c = wmma16(a, bw, c);
    }
    int col = nc*16 + m;
    float beta = s_batt[col];
    float vals[8]; float mx = -3.0e38f;
    #pragma unroll
    for (int r = 0; r < 8; ++r){ vals[r] = c[r] + beta; mx = fmaxf(mx, vals[r]); }
    mx = fmaxf(mx, __shfl_xor(mx, 16, 32));
    float se = 0.f;
    #pragma unroll
    for (int r = 0; r < 8; ++r){ vals[r] = __expf(vals[r] - mx); se += vals[r]; }
    se += __shfl_xor(se, 16, 32);
    float inv = 1.f / se;
    float acc = 0.f;
    #pragma unroll
    for (int r = 0; r < 8; ++r) acc += (float)fcat[(roff + r)*64 + col] * (vals[r] * inv);
    acc += __shfl_xor(acc, 16, 32);
    if (hiw == 0) s_agg[wave][col] = acc;
  }
  __syncthreads();

  // f_agg = lrelu(bn(agg @ Wc1 + bc1)) : lane = output channel (32)
  {
    int c = lane;
    float acc = p.bc1[c];
    const float* ag = s_agg[wave];
    #pragma unroll 8
    for (int k = 0; k < DOUT; ++k) acc += ag[k] * p.Wc1[k*DH + c];
    float s = p.ag2[c] * RS;
    acc = lrelu(acc*s + p.abe2[c]);
    p.f_agg[pt*DH + c] = (h16)acc;
  }
}

// ---------------------------------------------------------------------------
// Kernel 3: re-encode + attention pool 2 + output tail. One wave per point.
// ---------------------------------------------------------------------------
__global__ __launch_bounds__(256) void k_att2_out(P p){
  __shared__ h16 s_w2[32 * 32];
  __shared__ h16 s_w3[32 * 32];
  __shared__ h16 s_wf[DOUT * DOUT];
  __shared__ float s_b2[32], s_b3[32], s_batt[64];
  __shared__ h16 s_fx[8][16 * 32];    // f_xyz_e per wave (8KB)
  __shared__ h16 s_fcat[8][16 * 64];  // f_cat2 per wave (16KB)
  __shared__ float s_agg[8][64];
  __shared__ float s_fpc2[8][64];
  __shared__ float s_xv[8][32];

  const int tid = threadIdx.x;
  const float RS = rsqrtf(1.0f + EPSF);
  for (int i = tid; i < DOUT * DOUT; i += 256){
    int c = i / 64, k = i % 64;
    s_wf[c*64 + k] = (h16)(p.Wf2[k*64 + c] * (p.bg1[c] * RS));
  }
  for (int i = tid; i < 32 * 32; i += 256){
    int c = i / 32, k = i % 32;
    float w = (k < 10) ? p.W2[k*32 + c] : 0.f;
    s_w2[c*32 + k] = (h16)(w * (p.g2[c] * RS));
    s_w3[c*32 + k] = (h16)(p.W3[k*32 + c] * (p.g3[c] * RS));
  }
  for (int i = tid; i < 64; i += 256) s_batt[i] = p.bf2[i]*(p.bg1[i]*RS) + p.bbe1[i];
  for (int i = tid; i < 32; i += 256){
    s_b2[i] = p.b2[i]*(p.g2[i]*RS) + p.be2[i];
    s_b3[i] = p.b3[i]*(p.g3[i]*RS) + p.be3[i];
  }
  __syncthreads();

  const int wave = tid >> 5, lane = tid & 31;
  const int m = lane & 15, hiw = lane >> 4;
  const int kbase = hiw * 8, roff = hiw * 8;
  const long pt = (long)blockIdx.x * 8 + wave;
  const int b = (int)(pt / NN), n = (int)(pt % NN);

  const int idxm = p.nidx[pt*KNB + m];
  const float* xb = p.xyz + (long)b * 3 * NN;
  float cx = xb[n],    cy = xb[NN + n],    cz = xb[2*NN + n];
  float nx = xb[idxm], ny = xb[NN + idxm], nz = xb[2*NN + idxm];
  float rx = cx - nx, ry = cy - ny, rz = cz - nz;
  float d = sqrtf(rx*rx + ry*ry + rz*rz);
  float f10[10] = {d, rx, ry, rz, cx, cy, cz, nx, ny, nz};

  v16h ax;
  #pragma unroll
  for (int h = 0; h < 16; ++h) ax[h] = (h16)0.f;
  #pragma unroll
  for (int h = 0; h < 16; ++h){ int Kk = hiw*16 + h; if (Kk < 10) ax[h] = (h16)f10[Kk]; }

  h16* fx   = s_fx[wave];
  h16* fcat = s_fcat[wave];

  // f_xyz_e -> s_fx
  #pragma unroll
  for (int nc = 0; nc < 2; ++nc){
    int col = nc*16 + m;
    v16h bw = *(const v16h*)(s_w2 + col*32 + hiw*16);
    v8f c = {};
    c = wmma16(ax, bw, c);
    float beta = s_b2[col];
    #pragma unroll
    for (int r = 0; r < 8; ++r)
      fx[(roff + r)*32 + col] = (h16)lrelu(c[r] + beta);
  }

  // f_nb2 gather from f_agg -> fcat cols [0..31]
  {
    const h16* src = p.f_agg + ((long)b*NN + idxm)*DH + hiw*16;
    v8h a0 = *(const v8h*)src;
    v8h a1 = *(const v8h*)(src + 8);
    *(v8h*)(fcat + m*64 + hiw*16)     = a0;
    *(v8h*)(fcat + m*64 + hiw*16 + 8) = a1;
  }
  // shortcut input x[p][:]
  s_xv[wave][lane] = p.feature[((long)b*DIN + lane)*NN + n];
  __syncthreads();

  // f_xyz2 = lrelu(bn(f_xyz_e @ W3)) -> fcat cols [32..63]
  {
    v16h a = a_frag_lds(fx, 32, m, kbase, 0);
    #pragma unroll
    for (int nc = 0; nc < 2; ++nc){
      int col = nc*16 + m;
      v16h bw = *(const v16h*)(s_w3 + col*32 + hiw*16);
      v8f c = {};
      c = wmma16(a, bw, c);
      float beta = s_b3[col];
      #pragma unroll
      for (int r = 0; r < 8; ++r)
        fcat[(roff + r)*64 + 32 + col] = (h16)lrelu(c[r] + beta);
    }
  }
  __syncthreads();

  // attention pool 2
  #pragma unroll
  for (int nc = 0; nc < 4; ++nc){
    v8f c = {};
    #pragma unroll
    for (int kc = 0; kc < 2; ++kc){
      v16h a = a_frag_lds(fcat, 64, m, kbase, kc*32);
      v16h bw = *(const v16h*)(s_wf + (nc*16 + m)*DOUT + kc*32 + hiw*16);
      c = wmma16(a, bw, c);
    }
    int col = nc*16 + m;
    float beta = s_batt[col];
    float vals[8]; float mx = -3.0e38f;
    #pragma unroll
    for (int r = 0; r < 8; ++r){ vals[r] = c[r] + beta; mx = fmaxf(mx, vals[r]); }
    mx = fmaxf(mx, __shfl_xor(mx, 16, 32));
    float se = 0.f;
    #pragma unroll
    for (int r = 0; r < 8; ++r){ vals[r] = __expf(vals[r] - mx); se += vals[r]; }
    se += __shfl_xor(se, 16, 32);
    float inv = 1.f / se;
    float acc = 0.f;
    #pragma unroll
    for (int r = 0; r < 8; ++r) acc += (float)fcat[(roff + r)*64 + col] * (vals[r] * inv);
    acc += __shfl_xor(acc, 16, 32);
    if (hiw == 0) s_agg[wave][col] = acc;
  }
  __syncthreads();

  // f_pc2 = lrelu(bn(agg @ Wc2 + bc2)) : 64 outputs, 2 per lane
  #pragma unroll
  for (int t = 0; t < 2; ++t){
    int c = t*32 + lane;
    float acc = p.bc2[c];
    const float* ag = s_agg[wave];
    #pragma unroll 8
    for (int k = 0; k < DOUT; ++k) acc += ag[k] * p.Wc2[k*DOUT + c];
    float s = p.bg2[c] * RS;
    s_fpc2[wave][c] = lrelu(acc*s + p.bbe2[c]);
  }
  __syncthreads();

  // out = lrelu(bn(f_pc2 @ W4 + b4) + bn(x @ W5 + b5)), channel-major store
  #pragma unroll
  for (int t = 0; t < 4; ++t){
    int j = t*32 + lane;
    float mn = p.b4[j];
    const float* fp = s_fpc2[wave];
    #pragma unroll 8
    for (int k = 0; k < DOUT; ++k) mn += fp[k] * p.W4[k*128 + j];
    mn = mn * (p.g4[j] * RS) + p.be4[j];
    float sh = p.b5[j];
    const float* xv = s_xv[wave];
    #pragma unroll 8
    for (int k = 0; k < DIN; ++k) sh += xv[k] * p.W5[k*128 + j];
    sh = sh * (p.g5[j] * RS) + p.be5[j];
    p.out[((long)b*128 + j)*NN + n] = lrelu(mn + sh);
  }
}

// ---------------------------------------------------------------------------
extern "C" void kernel_launch(void* const* d_in, const int* in_sizes, int n_in,
                              void* d_out, int out_size, void* d_ws, size_t ws_size,
                              hipStream_t stream){
  (void)in_sizes; (void)n_in; (void)out_size; (void)ws_size;
  auto F = [&](int i){ return (const float*)d_in[i]; };
  P p;
  p.feature = F(0); p.xyz = F(1); p.nidx = (const int*)d_in[2];
  p.W1 = F(3);  p.b1 = F(4);  p.g1 = F(5);  p.be1 = F(6);
  p.W2 = F(7);  p.b2 = F(8);  p.g2 = F(9);  p.be2 = F(10);
  p.Wf1 = F(11); p.bf1 = F(12); p.ag1 = F(13); p.abe1 = F(14);
  p.Wc1 = F(15); p.bc1 = F(16); p.ag2 = F(17); p.abe2 = F(18);
  p.W3 = F(19); p.b3 = F(20); p.g3 = F(21); p.be3 = F(22);
  p.Wf2 = F(23); p.bf2 = F(24); p.bg1 = F(25); p.bbe1 = F(26);
  p.Wc2 = F(27); p.bc2 = F(28); p.bg2 = F(29); p.bbe2 = F(30);
  p.W4 = F(31); p.b4 = F(32); p.g4 = F(33); p.be4 = F(34);
  p.W5 = F(35); p.b5 = F(36); p.g5 = F(37); p.be5 = F(38);
  p.out = (float*)d_out;
  p.f_pc  = (h16*)d_ws;
  p.f_agg = (h16*)((char*)d_ws + (size_t)BB*NN*DH*sizeof(h16));

  // 163,840 points total
  k_fpc     <<<BB*NN/(16*8), 256, 0, stream>>>(p);  // 16 points per wave
  k_att1    <<<BB*NN/8,      256, 0, stream>>>(p);  // 1 point per wave
  k_att2_out<<<BB*NN/8,      256, 0, stream>>>(p);  // 1 point per wave
}
